// Box2FeatureGenerator_29901562315077
// MI455X (gfx1250) — matrix operand order repaired and checked by reading, not compiled, not run on hardware
//
#include <hip/hip_runtime.h>
#include <hip/hip_bf16.h>
#include <stdint.h>

typedef __attribute__((ext_vector_type(16))) __bf16 v16bf;
typedef __attribute__((ext_vector_type(8)))  float  v8f;

#define C_CH 256
#define H_SZ 256
#define W_SZ 256
#define B_SZ 4
#define N_BOX 32
#define HW   65536
#define BCHW 67108864
#define STAGE_N 3456           /* 32 ci * 6 * 18 halo tile elements */
#define STAGE_PF 14            /* ceil(3456/256) prefetch regs per thread */

static __device__ __forceinline__ unsigned short f2bf(float f) {
    unsigned u = __float_as_uint(f);
    unsigned r = u + 0x7FFFu + ((u >> 16) & 1u);   // round-to-nearest-even
    return (unsigned short)(r >> 16);
}

// ---------------------------------------------------------------------------
// Weight conversion: conv_w[kj][co][ci][ty][tx] f32 -> Wb[kj][co][tap][ci] bf16
// ---------------------------------------------------------------------------
__global__ void wconv_kernel(const float* __restrict__ cw,
                             unsigned short* __restrict__ wb, int total) {
    int e = blockIdx.x * 256 + threadIdx.x;
    if (e >= total) return;
    int ci  = e & 255;
    int tap = (e >> 8) % 9;
    int rem = e / (9 * 256);
    int co  = rem & 255;
    int kj  = rem >> 8;
    float v = cw[((kj * 256 + co) * 256 + ci) * 9 + tap];
    wb[e] = f2bf(v);
}

// ---------------------------------------------------------------------------
// MLP: obj(25) -> 256 -> 256 -> 256, scaled by score. One block per object.
// ---------------------------------------------------------------------------
__global__ void mlp_kernel(const float* __restrict__ pb, const float* __restrict__ ps,
                           const float* __restrict__ W1, const float* __restrict__ b1,
                           const float* __restrict__ W2, const float* __restrict__ b2,
                           const float* __restrict__ W3, const float* __restrict__ b3,
                           float* __restrict__ objf) {
    __shared__ float sobj[25];
    __shared__ float h1[256];
    __shared__ float h2[256];
    int o = blockIdx.x;
    int c = threadIdx.x;
    if (c < 24) sobj[c] = pb[o * 24 + c];
    if (c == 24) sobj[24] = ps[o];
    __syncthreads();
    float acc = b1[c];
    #pragma unroll
    for (int i = 0; i < 25; ++i) acc += sobj[i] * W1[i * 256 + c];
    h1[c] = fmaxf(acc, 0.f);
    __syncthreads();
    acc = b2[c];
    for (int i = 0; i < 256; ++i) acc += h1[i] * W2[i * 256 + c];
    h2[c] = fmaxf(acc, 0.f);
    __syncthreads();
    acc = b3[c];
    for (int i = 0; i < 256; ++i) acc += h2[i] * W3[i * 256 + c];
    objf[o * 256 + c] = acc * ps[o];
}

// ---------------------------------------------------------------------------
// Per-box grid-aligned bounding boxes
// ---------------------------------------------------------------------------
__global__ void bbox_kernel(const float* __restrict__ pb, int4* __restrict__ bbox) {
    int t = threadIdx.x;
    if (t >= B_SZ * N_BOX) return;
    float mnx = 1e30f, mxx = -1e30f, mny = 1e30f, mxy = -1e30f;
    #pragma unroll
    for (int j = 0; j < 8; ++j) {
        float px = pb[t * 24 + j * 3 + 0];
        float py = pb[t * 24 + j * 3 + 1];
        float gx = floorf((px + 51.2f) * 2.5f);   // /0.4
        float gy = floorf((py + 51.2f) * 2.5f);
        mnx = fminf(mnx, gx); mxx = fmaxf(mxx, gx);
        mny = fminf(mny, gy); mxy = fmaxf(mxy, gy);
    }
    int4 v;
    v.x = (int)fminf(fmaxf(mnx, 0.f), 255.f);
    v.y = (int)fminf(fmaxf(mxx, 0.f), 255.f);
    v.z = (int)fminf(fmaxf(mny, 0.f), 255.f);
    v.w = (int)fminf(fmaxf(mxy, 0.f), 255.f);
    bbox[t] = v;
}

// ---------------------------------------------------------------------------
// Rasterize: last (highest-index) covering box per pixel
// ---------------------------------------------------------------------------
__global__ void raster_kernel(const int4* __restrict__ bbox, int* __restrict__ idxmap) {
    __shared__ int4 bb[N_BOX];
    int b = blockIdx.x >> 8;
    int y = blockIdx.x & 255;
    int x = threadIdx.x;
    if (x < N_BOX) bb[x] = bbox[b * N_BOX + x];
    __syncthreads();
    int last = -1;
    #pragma unroll
    for (int n = 0; n < N_BOX; ++n) {
        int4 v = bb[n];
        bool cov = (x >= v.x) & (x <= v.y) & (y >= v.z) & (y <= v.w);
        last = cov ? n : last;
    }
    idxmap[b * HW + y * 256 + x] = last;
}

// ---------------------------------------------------------------------------
// Gather features into NCHW f32 (writes initial activation into d_out)
// ---------------------------------------------------------------------------
__global__ void gather_kernel(const float* __restrict__ objf,
                              const int* __restrict__ idxmap,
                              float* __restrict__ out) {
    int e = blockIdx.x * 256 + threadIdx.x;
    int b = e >> 24;
    int c = (e >> 16) & 255;
    int p = e & 65535;
    int last = idxmap[b * HW + p];
    out[e] = (last >= 0) ? objf[(b * N_BOX + last) * 256 + c] : 0.f;
}

// ---------------------------------------------------------------------------
// Staging helpers for the conv kernel (software-pipelined, optional fused BN)
// ---------------------------------------------------------------------------
template <bool BN>
__device__ __forceinline__ void stage_load(const float* __restrict__ in, int b,
                                           int ci_base, int x0, int y0, int tid,
                                           const float* __restrict__ scale,
                                           const float* __restrict__ shift,
                                           float pf[STAGE_PF]) {
    #pragma unroll
    for (int j = 0; j < STAGE_PF; ++j) {
        int i = tid + j * 256;
        float v = 0.f;
        if (i < STAGE_N) {
            int ci  = i / 108;
            int rem = i - ci * 108;
            int yy  = rem / 18;
            int xx  = rem - yy * 18;
            int gy = y0 - 1 + yy;
            int gx = x0 - 1 + xx;
            if ((unsigned)gy < 256u && (unsigned)gx < 256u)
                v = in[((b * 256 + ci_base + ci) * 256 + gy) * 256 + gx];
            if (BN) {
                int cc = b * 256 + ci_base + ci;
                v = fmaxf(v * scale[cc] + shift[cc], 0.f);   // BN + ReLU on the fly
            }
        }
        pf[j] = v;
    }
}

__device__ __forceinline__ void stage_store(unsigned short* __restrict__ tile,
                                            int tid, const float pf[STAGE_PF]) {
    #pragma unroll
    for (int j = 0; j < STAGE_PF; ++j) {
        int i = tid + j * 256;
        if (i < STAGE_N) {
            int ci  = i / 108;
            int rem = i - ci * 108;
            int yy  = rem / 18;
            int xx  = rem - yy * 18;
            tile[(yy * 18 + xx) * 32 + ci] = f2bf(pf[j]);
        }
    }
}

// ---------------------------------------------------------------------------
// Implicit-GEMM 3x3 conv with bf16 WMMA + fused BN partial stats.
// block: 16x4 pixel tile x all 256 out-channels; 8 waves, wave w -> co [32w,32w+32)
// LDS tile double-buffered; next ci-block's global loads overlap the WMMAs.
// ---------------------------------------------------------------------------
union ABFrag { v16bf v; unsigned u[8]; uint4 q[2]; };

template <bool BN>
__global__ __launch_bounds__(256)
void conv_kernel(const float* __restrict__ in, float* __restrict__ out,
                 const unsigned short* __restrict__ wb,
                 float* __restrict__ ssum, float* __restrict__ ssq,
                 const float* __restrict__ scale, const float* __restrict__ shift) {
    __shared__ unsigned short tile[2][STAGE_N];    // [buf][ (y*18+x)*32 + ci ], bf16
    const int tid  = threadIdx.x;
    const int wave = tid >> 5;
    const int lane = tid & 31;
    const int half = lane >> 4;
    const int n    = lane & 15;
    const int x0 = blockIdx.x * 16;
    const int y0 = blockIdx.y * 4;
    const int b  = blockIdx.z;
    const int co_base = wave * 32;

    v8f acc[2][4];
    #pragma unroll
    for (int cj = 0; cj < 2; ++cj)
        #pragma unroll
        for (int pg = 0; pg < 4; ++pg)
            #pragma unroll
            for (int e = 0; e < 8; ++e) acc[cj][pg][e] = 0.f;

    float pf[STAGE_PF];
    stage_load<BN>(in, b, 0, x0, y0, tid, scale, shift, pf);
    stage_store(tile[0], tid, pf);
    __syncthreads();

    for (int cb = 0; cb < 8; ++cb) {
        const int ci_base = cb * 32;
        if (cb < 7)   // prefetch next ci-block; latency hidden by the WMMAs below
            stage_load<BN>(in, b, ci_base + 32, x0, y0, tid, scale, shift, pf);

        const unsigned* tp32 = (const unsigned*)tile[cb & 1];

        #pragma unroll
        for (int tap = 0; tap < 9; ++tap) {
            const int dy = tap / 3, dx = tap - 3 * (tap / 3);
            // A fragments (weights) straight from global (hot in L2)
            ABFrag a[2];
            #pragma unroll
            for (int cj = 0; cj < 2; ++cj) {
                const int co = co_base + cj * 16 + n;
                const unsigned short* wr = wb + (co * 9 + tap) * 256 + ci_base;
                a[cj].q[0] = *(const uint4*)(wr + 8 * half);        // K = 8h..8h+7
                a[cj].q[1] = *(const uint4*)(wr + 16 + 8 * half);   // K = 16+8h..
            }
            #pragma unroll
            for (int pg = 0; pg < 4; ++pg) {
                ABFrag bf;
                const int yy = pg + dy;
                const int xx = n + dx;
                const unsigned base = (unsigned)((yy * 18 + xx) * 16 + half * 4);
                #pragma unroll
                for (int r = 0; r < 8; ++r) bf.u[r] = tp32[base + (r & 3) + (r >> 2) * 8]; // K=8h+2r (r<4), 16+8h+2(r-4) (r>=4)
                acc[0][pg] = __builtin_amdgcn_wmma_f32_16x16x32_bf16(
                    false, a[0].v, false, bf.v, (short)0, acc[0][pg], false, false);
                acc[1][pg] = __builtin_amdgcn_wmma_f32_16x16x32_bf16(
                    false, a[1].v, false, bf.v, (short)0, acc[1][pg], false, false);
            }
        }

        if (cb < 7)
            stage_store(tile[(cb + 1) & 1], tid, pf);
        __syncthreads();
    }

    // Store results + per-channel BN partial stats
    #pragma unroll
    for (int cj = 0; cj < 2; ++cj) {
        v8f s, sq;
        #pragma unroll
        for (int e = 0; e < 8; ++e) { s[e] = 0.f; sq[e] = 0.f; }
        #pragma unroll
        for (int pg = 0; pg < 4; ++pg) {
            #pragma unroll
            for (int v = 0; v < 8; ++v) {
                float val = acc[cj][pg][v];
                int co = co_base + cj * 16 + half * 8 + v;
                out[((b * 256 + co) * 256 + (y0 + pg)) * 256 + x0 + n] = val;
                s[v] += val; sq[v] += val * val;
            }
        }
        #pragma unroll
        for (int m = 1; m < 16; m <<= 1)
            #pragma unroll
            for (int v = 0; v < 8; ++v) {
                s[v]  += __shfl_xor(s[v],  m, 32);
                sq[v] += __shfl_xor(sq[v], m, 32);
            }
        if (n == 0) {
            #pragma unroll
            for (int v = 0; v < 8; ++v) {
                int co = co_base + cj * 16 + half * 8 + v;
                atomicAdd(&ssum[b * 256 + co], s[v]);
                atomicAdd(&ssq [b * 256 + co], sq[v]);
            }
        }
    }
}

// ---------------------------------------------------------------------------
// BN stats helpers + residual elementwise pass
// stats layout: [0,1024)=sum  [1024,2048)=sumsq  [2048,3072)=scale  [3072,4096)=shift
// ---------------------------------------------------------------------------
__global__ void zero_stats_kernel(float* __restrict__ st) {
    int t = blockIdx.x * 256 + threadIdx.x;
    if (t < 2048) st[t] = 0.f;
}

__global__ void finalize_kernel(float* __restrict__ st,
                                const float* __restrict__ gam,
                                const float* __restrict__ bet) {
    int t = blockIdx.x * 256 + threadIdx.x;
    if (t < 1024) {
        int c = t & 255;
        float mean = st[t] * (1.f / 65536.f);
        float ex2  = st[1024 + t] * (1.f / 65536.f);
        float istd = rsqrtf(ex2 - mean * mean + 1e-5f);
        float sc = istd * gam[c];
        st[2048 + t] = sc;                  // scale
        st[3072 + t] = bet[c] - mean * sc;  // shift
    }
}

__global__ void bnres_kernel(const float* __restrict__ y2, const float* __restrict__ st,
                             float* __restrict__ x) {
    int e = blockIdx.x * 256 + threadIdx.x;
    int b = e >> 24;
    int c = (e >> 16) & 255;
    float sc = st[2048 + b * 256 + c];
    float sh = st[3072 + b * 256 + c];
    float v = y2[e] * sc + sh + x[e];
    x[e] = fmaxf(v, 0.f);
}

// ---------------------------------------------------------------------------
extern "C" void kernel_launch(void* const* d_in, const int* in_sizes, int n_in,
                              void* d_out, int out_size, void* d_ws, size_t ws_size,
                              hipStream_t stream) {
    const float* pb  = (const float*)d_in[0];   // (4,32,8,3)
    const float* ps  = (const float*)d_in[1];   // (4,32)
    const float* W1  = (const float*)d_in[2];
    const float* b1  = (const float*)d_in[3];
    const float* W2  = (const float*)d_in[4];
    const float* b2  = (const float*)d_in[5];
    const float* W3  = (const float*)d_in[6];
    const float* b3  = (const float*)d_in[7];
    const float* cw  = (const float*)d_in[8];   // (3,2,256,256,3,3)
    const float* gam = (const float*)d_in[9];   // (3,2,256)
    const float* bet = (const float*)d_in[10];  // (3,2,256)
    float* xbuf = (float*)d_out;                // activation ping buffer (NCHW)

    char* ws = (char*)d_ws;
    const size_t WB_OFF   = 0;                  // 6*256*9*256 bf16 = 7,077,888 B
    const size_t OBJ_OFF  = 7077888;            // 128*256 f32
    const size_t BBOX_OFF = OBJ_OFF + 131072;   // 128 int4
    const size_t IDX_OFF  = BBOX_OFF + 2048;    // 4*65536 int
    const size_t ST_OFF   = IDX_OFF + 1048576;  // 4096 f32 (sum|sumsq|scale|shift)
    const size_t Y_OFF    = ST_OFF + 16384;     // 268,435,456 B
    const size_t Z_OFF    = Y_OFF + 268435456;  // 268,435,456 B

    unsigned short* wbuf = (unsigned short*)(ws + WB_OFF);
    float* objf  = (float*)(ws + OBJ_OFF);
    int4*  bbox  = (int4*)(ws + BBOX_OFF);
    int*   idxm  = (int*)(ws + IDX_OFF);
    float* stats = (float*)(ws + ST_OFF);
    float* ybuf  = (float*)(ws + Y_OFF);
    float* zbuf  = (float*)(ws + Z_OFF);

    const int wtotal = 6 * 256 * 9 * 256;
    wconv_kernel<<<(wtotal + 255) / 256, 256, 0, stream>>>(cw, wbuf, wtotal);
    mlp_kernel<<<B_SZ * N_BOX, 256, 0, stream>>>(pb, ps, W1, b1, W2, b2, W3, b3, objf);
    bbox_kernel<<<1, 128, 0, stream>>>(pb, bbox);
    raster_kernel<<<B_SZ * 256, 256, 0, stream>>>(bbox, idxm);
    gather_kernel<<<BCHW / 256, 256, 0, stream>>>(objf, idxm, xbuf);

    dim3 cgrid(W_SZ / 16, H_SZ / 4, B_SZ);
    const int egrid = BCHW / 256;
    float* sc = stats + 2048;
    float* sh = stats + 3072;
    for (int k = 0; k < 3; ++k) {
        const unsigned short* w0 = wbuf + (size_t)(k * 2 + 0) * 256 * 9 * 256;
        const unsigned short* w1 = wbuf + (size_t)(k * 2 + 1) * 256 * 9 * 256;
        const float* g0 = gam + (k * 2 + 0) * 256; const float* be0 = bet + (k * 2 + 0) * 256;
        const float* g1 = gam + (k * 2 + 1) * 256; const float* be1 = bet + (k * 2 + 1) * 256;

        // conv1: plain input, raw conv output + stats
        zero_stats_kernel<<<8, 256, 0, stream>>>(stats);
        conv_kernel<false><<<cgrid, 256, 0, stream>>>(xbuf, ybuf, w0,
                                                      stats, stats + 1024, sc, sh);
        finalize_kernel<<<4, 256, 0, stream>>>(stats, g0, be0);

        // conv2: BN+ReLU of y1 fused into input staging; raw conv output + stats
        zero_stats_kernel<<<8, 256, 0, stream>>>(stats);
        conv_kernel<true><<<cgrid, 256, 0, stream>>>(ybuf, zbuf, w1,
                                                     stats, stats + 1024, sc, sh);
        finalize_kernel<<<4, 256, 0, stream>>>(stats, g1, be1);

        // x = relu(bn(y2) + x)
        bnres_kernel<<<egrid, 256, 0, stream>>>(zbuf, stats, xbuf);
    }
}
